// LSTMModel_56813827391752
// MI455X (gfx1250) — matrix operand order repaired and checked by reading, not compiled, not run on hardware
//
#include <hip/hip_runtime.h>
#include <hip/hip_bf16.h>

#define B_   64
#define S_   1024
#define D_   256
#define H_   32
#define V_   2048
#define G4H  128          // 4*H

typedef _Float16 v8h  __attribute__((ext_vector_type(8)));
typedef _Float16 v16h __attribute__((ext_vector_type(16)));
typedef float    v8f  __attribute__((ext_vector_type(8)));

__device__ __forceinline__ v8f wmma_f32_f16(v16h a, v16h b, v8f c) {
    // 8 args: (neg_a, A, neg_b, B, c_mod, C, reuse_a, reuse_b)
    return __builtin_amdgcn_wmma_f32_16x16x32_f16(false, a, false, b, (short)0, c, false, false);
}

// A-fragment (16x32 f16) from row-major [16][stride] f16 buffer.
// lane<16: row=lane, K=0..15 ; lane>=16: row=lane-16, K=16..31
__device__ __forceinline__ v16h load_afrag(const _Float16* X, int stride, int kOff, int lane) {
    int row = lane & 15;
    const _Float16* p = X + row * stride + kOff + ((lane & 16) ? 16 : 0);
    v8h a0 = *(const v8h*)p;
    v8h a1 = *(const v8h*)(p + 8);
    v16h a;
#pragma unroll
    for (int i = 0; i < 8; i++) { a[i] = a0[i]; a[i + 8] = a1[i]; }
    return a;
}

__device__ __forceinline__ float sigf(float x) { return 1.0f / (1.0f + __expf(-x)); }

// ---------------- kernel 0: f32->f16 weight/embedding conversion + bias fold ----------
__global__ void convert_kernel(const float* emb, const float* Wih1, const float* Whh1,
                               const float* Wih2, const float* Whh2, const float* Wfc,
                               const float* bih1, const float* bhh1,
                               const float* bih2, const float* bhh2,
                               _Float16* emb16, _Float16* Wih1_16, _Float16* Whh1_16,
                               _Float16* Wih2_16, _Float16* Whh2_16, _Float16* Wfc16,
                               float* bias1, float* bias2) {
    int i = blockIdx.x * 256 + threadIdx.x;
    if (i < V_ * D_)  emb16[i]   = (_Float16)emb[i];
    if (i < G4H * D_) Wih1_16[i] = (_Float16)Wih1[i];
    if (i < G4H * H_) {
        Whh1_16[i] = (_Float16)Whh1[i];
        Wih2_16[i] = (_Float16)Wih2[i];
        Whh2_16[i] = (_Float16)Whh2[i];
    }
    if (i < V_ * H_)  Wfc16[i]   = (_Float16)Wfc[i];
    if (i < G4H) { bias1[i] = bih1[i] + bhh1[i]; bias2[i] = bih2[i] + bhh2[i]; }
}

// ---------------- kernel 1: G1[s*B+b, 0..127] = emb[x_ids] @ Wih1^T + bias1 ----------
// grid 4096 blocks (one 16-row M tile each), 8 waves = 8 N-tiles, K=256 -> 8 WMMAs/wave
__global__ void gates1_kernel(const int* __restrict__ x_ids, const _Float16* __restrict__ emb16,
                              const _Float16* __restrict__ Wih1_16,
                              const float* __restrict__ bias1, float* __restrict__ G1) {
    __shared__ __align__(16) _Float16 xs[16 * D_];
    const int r0  = blockIdx.x * 16;    // global row = s*B + b
    const int tid = threadIdx.x;

    {   // gather+stage 16 embedding rows (f16) into LDS
        int row = tid >> 4;             // 0..15
        int chunk = (tid & 15) * 16;    // 16 f16 per thread
        int g = r0 + row;
        int s = g >> 6;                 // /B
        int b = g & 63;
        int id = x_ids[b * S_ + s];
        const v8h* src = (const v8h*)(emb16 + (size_t)id * D_ + chunk);
        v8h* dst = (v8h*)(xs + row * D_ + chunk);
        dst[0] = src[0];
        dst[1] = src[1];
    }
    __syncthreads();

    const int w = tid >> 5, lane = tid & 31;
    const int col = w * 16 + (lane & 15);
    const float bv = bias1[col];
    v8f acc;
#pragma unroll
    for (int i = 0; i < 8; i++) acc[i] = bv;

#pragma unroll
    for (int kb = 0; kb < 8; kb++) {
        v16h a = load_afrag(xs, D_, kb * 32, lane);
        const _Float16* bp = Wih1_16 + (size_t)col * D_ + kb * 32 + ((lane & 16) ? 16 : 0);
        v16h bf = *(const v16h*)bp;
        acc = wmma_f32_f16(a, bf, acc);
    }
    const int mBase = r0 + ((lane & 16) ? 8 : 0);
#pragma unroll
    for (int r = 0; r < 8; r++)
        G1[(size_t)(mBase + r) * G4H + col] = acc[r];
}

// ---------------- kernel 2: sequential 2-layer LSTM recurrence ----------------------
// 4 blocks x 256 threads; block owns batch rows [16*blk, 16*blk+16); 8 waves = 8 gate N-tiles
__global__ void lstm_rec_kernel(const float* __restrict__ G1,
                                const _Float16* __restrict__ Whh1_16,
                                const _Float16* __restrict__ Wih2_16,
                                const _Float16* __restrict__ Whh2_16,
                                const float* __restrict__ bias2,
                                float* __restrict__ H2) {
    __shared__ __align__(16) _Float16 h1f[16 * H_];
    __shared__ __align__(16) _Float16 h2f[16 * H_];
    __shared__ float c1s[16 * H_];
    __shared__ float c2s[16 * H_];
    __shared__ float gts[16 * G4H];

    const int tid = threadIdx.x;
    const int w = tid >> 5, lane = tid & 31;
    const int b0 = blockIdx.x * 16;
    const int col = w * 16 + (lane & 15);
    const int hi = (lane & 16) ? 1 : 0;
    const int mOff = hi ? 8 : 0;

    // persistent weight B-fragments (K=32 -> one v16h per lane per matrix)
    v16h Bhh1 = *(const v16h*)(Whh1_16 + col * H_ + (hi ? 16 : 0));
    v16h Bih2 = *(const v16h*)(Wih2_16 + col * H_ + (hi ? 16 : 0));
    v16h Bhh2 = *(const v16h*)(Whh2_16 + col * H_ + (hi ? 16 : 0));
    const float b2v = bias2[col];

    for (int e = tid; e < 16 * H_; e += 256) {
        c1s[e] = 0.f; c2s[e] = 0.f;
        h1f[e] = (_Float16)0.f; h2f[e] = (_Float16)0.f;
    }
    __syncthreads();

    for (int t = 0; t < S_; t++) {
        const size_t gbase = (size_t)t * B_ + b0;
        // prefetch next step's precomputed gate tile (8KB/block) into caches
        if (t + 1 < S_)
            __builtin_prefetch(G1 + (gbase + B_) * G4H + tid * 8, 0, 0);

        // ---- layer 1: acc = G1_tile + h1 @ Whh1^T
        {
            v8f acc;
#pragma unroll
            for (int r = 0; r < 8; r++)
                acc[r] = G1[(gbase + mOff + r) * G4H + col];
            v16h a = load_afrag(h1f, H_, 0, lane);
            acc = wmma_f32_f16(a, Bhh1, acc);
#pragma unroll
            for (int r = 0; r < 8; r++)
                gts[(mOff + r) * G4H + col] = acc[r];
        }
        __syncthreads();
        // ---- elementwise layer 1
#pragma unroll
        for (int q = 0; q < 2; q++) {
            int e = tid + q * 256;
            int row = e >> 5, c = e & 31;
            float gi = gts[row * G4H + c];
            float gf = gts[row * G4H + 32 + c];
            float gg = gts[row * G4H + 64 + c];
            float go = gts[row * G4H + 96 + c];
            float cn = sigf(gf) * c1s[e] + sigf(gi) * tanhf(gg);
            c1s[e] = cn;
            h1f[e] = (_Float16)(sigf(go) * tanhf(cn));
        }
        __syncthreads();
        // ---- layer 2: acc = bias2 + h1 @ Wih2^T + h2 @ Whh2^T
        {
            v8f acc;
#pragma unroll
            for (int r = 0; r < 8; r++) acc[r] = b2v;
            v16h a1 = load_afrag(h1f, H_, 0, lane);
            acc = wmma_f32_f16(a1, Bih2, acc);
            v16h a2 = load_afrag(h2f, H_, 0, lane);
            acc = wmma_f32_f16(a2, Bhh2, acc);
#pragma unroll
            for (int r = 0; r < 8; r++)
                gts[(mOff + r) * G4H + col] = acc[r];
        }
        __syncthreads();
        // ---- elementwise layer 2 + H2 store
#pragma unroll
        for (int q = 0; q < 2; q++) {
            int e = tid + q * 256;
            int row = e >> 5, c = e & 31;
            float gi = gts[row * G4H + c];
            float gf = gts[row * G4H + 32 + c];
            float gg = gts[row * G4H + 64 + c];
            float go = gts[row * G4H + 96 + c];
            float cn = sigf(gf) * c2s[e] + sigf(gi) * tanhf(gg);
            c2s[e] = cn;
            float hn = sigf(go) * tanhf(cn);
            h2f[e] = (_Float16)hn;
            H2[(gbase + row) * H_ + c] = hn;
        }
        __syncthreads();
    }
}

// ---------------- kernel 3: out = softmax(H2 @ Wfc^T + b_fc), NT-stored -------------
// grid 4096 blocks (16-row M tile), 8 waves x 16 N-tiles = 2048 cols; logits in regs
__global__ void fc_softmax_kernel(const float* __restrict__ H2,
                                  const _Float16* __restrict__ Wfc16,
                                  const float* __restrict__ bfc,
                                  float* __restrict__ out) {
    __shared__ float redbuf[8 * 16];
    __shared__ float rowred[16];
    const int tid = threadIdx.x;
    const int w = tid >> 5, lane = tid & 31;
    const int r0 = blockIdx.x * 16;
    const int hi = (lane & 16) ? 1 : 0;
    const int l15 = lane & 15;

    // A fragment: H2 rows (f32 -> f16)
    v16h a;
    {
        const float* p = H2 + (size_t)(r0 + l15) * H_ + (hi ? 16 : 0);
#pragma unroll
        for (int i = 0; i < 8; i++) { a[i] = (_Float16)p[i]; a[i + 8] = (_Float16)p[i + 8]; }
    }

    v8f acc[16];
#pragma unroll
    for (int t = 0; t < 16; t++) {
        int n = w * 256 + t * 16 + l15;
        v16h bf = *(const v16h*)(Wfc16 + n * H_ + (hi ? 16 : 0));
        float bb = bfc[n];
        v8f c;
#pragma unroll
        for (int r = 0; r < 8; r++) c[r] = bb;
        acc[t] = wmma_f32_f16(a, bf, c);
    }

    // ---- row max (this lane holds rows hi*8 .. hi*8+7 across its 16 cols)
    float pm[8];
#pragma unroll
    for (int r = 0; r < 8; r++) {
        float m = acc[0][r];
#pragma unroll
        for (int t = 1; t < 16; t++) m = fmaxf(m, acc[t][r]);
        pm[r] = m;
    }
#pragma unroll
    for (int off = 1; off < 16; off <<= 1)
#pragma unroll
        for (int r = 0; r < 8; r++) pm[r] = fmaxf(pm[r], __shfl_xor(pm[r], off));
    if (l15 == 0) {
#pragma unroll
        for (int r = 0; r < 8; r++) redbuf[w * 16 + hi * 8 + r] = pm[r];
    }
    __syncthreads();
    if (tid < 16) {
        float m = redbuf[tid];
        for (int ww = 1; ww < 8; ww++) m = fmaxf(m, redbuf[ww * 16 + tid]);
        rowred[tid] = m;
    }
    __syncthreads();
    float mrow[8];
#pragma unroll
    for (int r = 0; r < 8; r++) mrow[r] = rowred[hi * 8 + r];

    // ---- row sum of exp (redbuf safe to reuse: all prior reads done before last barrier)
    float ps[8];
#pragma unroll
    for (int r = 0; r < 8; r++) {
        float s = 0.f;
#pragma unroll
        for (int t = 0; t < 16; t++) s += __expf(acc[t][r] - mrow[r]);
        ps[r] = s;
    }
#pragma unroll
    for (int off = 1; off < 16; off <<= 1)
#pragma unroll
        for (int r = 0; r < 8; r++) ps[r] += __shfl_xor(ps[r], off);
    if (l15 == 0) {
#pragma unroll
        for (int r = 0; r < 8; r++) redbuf[w * 16 + hi * 8 + r] = ps[r];
    }
    __syncthreads();
    if (tid < 16) {
        float s = 0.f;
        for (int ww = 0; ww < 8; ww++) s += redbuf[ww * 16 + tid];
        rowred[tid] = s;
    }
    __syncthreads();
    float inv[8];
#pragma unroll
    for (int r = 0; r < 8; r++) inv[r] = 1.f / rowred[hi * 8 + r];

    // ---- write normalized probabilities (streaming 512MB -> non-temporal)
#pragma unroll
    for (int r = 0; r < 8; r++) {
        int g = r0 + hi * 8 + r;
        int s = g >> 6;   // /B
        int b = g & 63;
        float* orow = out + ((size_t)b * S_ + s) * V_;
#pragma unroll
        for (int t = 0; t < 16; t++) {
            int n = w * 256 + t * 16 + l15;
            __builtin_nontemporal_store(__expf(acc[t][r] - mrow[r]) * inv[r], orow + n);
        }
    }
}

// ---------------------------------- launcher ----------------------------------------
extern "C" void kernel_launch(void* const* d_in, const int* in_sizes, int n_in,
                              void* d_out, int out_size, void* d_ws, size_t ws_size,
                              hipStream_t stream) {
    (void)in_sizes; (void)n_in; (void)out_size; (void)ws_size;
    const int*   x_ids = (const int*)  d_in[0];
    const float* emb   = (const float*)d_in[1];
    const float* Wih1  = (const float*)d_in[2];
    const float* Whh1  = (const float*)d_in[3];
    const float* bih1  = (const float*)d_in[4];
    const float* bhh1  = (const float*)d_in[5];
    const float* Wih2  = (const float*)d_in[6];
    const float* Whh2  = (const float*)d_in[7];
    const float* bih2  = (const float*)d_in[8];
    const float* bhh2  = (const float*)d_in[9];
    const float* Wfc   = (const float*)d_in[10];
    const float* bfc   = (const float*)d_in[11];
    float* out = (float*)d_out;

    char* ws = (char*)d_ws;
    _Float16* emb16   = (_Float16*)(ws + 0);          // 1,048,576 B
    _Float16* Wih1_16 = (_Float16*)(ws + 1048576);    //    65,536 B
    _Float16* Whh1_16 = (_Float16*)(ws + 1114112);    //     8,192 B
    _Float16* Wih2_16 = (_Float16*)(ws + 1122304);    //     8,192 B
    _Float16* Whh2_16 = (_Float16*)(ws + 1130496);    //     8,192 B
    _Float16* Wfc16   = (_Float16*)(ws + 1138688);    //   131,072 B
    float*    bias1   = (float*)   (ws + 1269760);    //       512 B
    float*    bias2   = (float*)   (ws + 1270272);    //       512 B
    float*    G1      = (float*)   (ws + 1270784);    // 33,554,432 B  [S*B,128]
    float*    H2      = (float*)   (ws + 34825216);   //  8,388,608 B  [S*B,32]

    convert_kernel<<<(V_ * D_) / 256, 256, 0, stream>>>(
        emb, Wih1, Whh1, Wih2, Whh2, Wfc, bih1, bhh1, bih2, bhh2,
        emb16, Wih1_16, Whh1_16, Wih2_16, Whh2_16, Wfc16, bias1, bias2);

    gates1_kernel<<<(B_ * S_) / 16, 256, 0, stream>>>(x_ids, emb16, Wih1_16, bias1, G1);

    lstm_rec_kernel<<<B_ / 16, 256, 0, stream>>>(G1, Whh1_16, Wih2_16, Whh2_16, bias2, H2);

    fc_softmax_kernel<<<(B_ * S_) / 16, 256, 0, stream>>>(H2, Wfc16, bfc, out);
}
